// AlignmentModule_82248623718826
// MI455X (gfx1250) — matrix operand order repaired and checked by reading, not compiled, not run on hardware
//
#include <hip/hip_runtime.h>

// ---------------------------------------------------------------------------
// Types / helpers
// ---------------------------------------------------------------------------
typedef __attribute__((ext_vector_type(16))) __bf16 v16bf;
typedef __attribute__((ext_vector_type(8)))  float  v8f;

#define DEV static __device__ __forceinline__

DEV unsigned short f2bfu(float f) {
  unsigned u = __builtin_bit_cast(unsigned, f);
  unsigned r = (u + 0x7FFFu + ((u >> 16) & 1u)) >> 16;
  return (unsigned short)r;
}
DEV __bf16 f2bf(float f) { return __builtin_bit_cast(__bf16, f2bfu(f)); }
DEV __bf16 u2bf(unsigned short s) { return __builtin_bit_cast(__bf16, s); }

enum EpiBits {
  E_F32 = 1, E_BF16 = 2, E_BIAS = 4, E_RELU = 8, E_GELU = 16,
  E_ACCUM = 32, E_ABS = 64, E_MEAN = 128
};

// Problem constants
#define BB  128
#define LI  576
#define LT  77
#define DD  512

// d_out element offsets (fp32)
#define OFF_FIMG    0
#define OFF_FTXT    65536
#define OFF_LPT     131072
#define OFF_LREC    131073
#define OFF_RESIMG  131074
#define OFF_RESTXT  196610

// ---------------------------------------------------------------------------
// Generic bf16 WMMA GEMM:  C[M,N] = epilogue( A[M,K] * B[K,N] )
//   AKM : A stored [K][M] row-major -> transposed while staging
//   BKN : B stored [K][N] row-major (weights) -> transposed while staging
//         else B stored [N][K] row-major (A*B^T form)
// Block: 256 threads (8 waves). Block tile 64(M) x 128(N) x 32(K).
// Wave tile 32x32 = 2x2 WMMA frags (4 v_wmma per K step).
// Ping-pong LDS double buffering: one barrier per K step; next tile's
// global loads are issued before this tile's fragment reads.
// ---------------------------------------------------------------------------
template <int E, bool AKM, bool BKN>
__global__ __launch_bounds__(256) void gemm_wmma(
    const __bf16* __restrict__ A, int lda, long sA,
    const __bf16* __restrict__ Bm, int ldb, long sB,
    float* Cf, __bf16* Cbf, int ldc, long sC,
    const float* __restrict__ bias,
    const float* __restrict__ ref, long sRef,
    float* redAcc, float invL,
    int M, int N, int K) {
  __shared__ __align__(16) __bf16 As[2][64][40];    // [m][k]
  __shared__ __align__(16) __bf16 Bs[2][128][40];   // [n][k] (transposed panel)

  const int tid = threadIdx.x;
  const int b   = blockIdx.z;
  const int bm  = blockIdx.y * 64;
  const int bn  = blockIdx.x * 128;
  A  += (long)b * sA;
  Bm += (long)b * sB;

  const int lane = tid & 31;
  const int wave = tid >> 5;
  const int lh   = lane >> 4;        // lane half
  const int r    = lane & 15;
  const int wn   = (wave & 3) * 32;  // wave N offset in tile
  const int wm   = (wave >> 2) * 32; // wave M offset in tile

  union U8 { uint4 v; unsigned short s[8]; };
  U8 ra, rb0, rb1;

  // ---- staging: one 16B run per thread per panel-half ----
  auto loadA = [&](int kb) {
    if (!AKM) {
      int mg = bm + (tid >> 2);
      int kg = kb + ((tid & 3) << 3);
      if (mg < M && (kg + 8) <= K) {
        ra.v = *(const uint4*)(A + (long)mg * lda + kg);
      } else {
#pragma unroll
        for (int i = 0; i < 8; ++i) {
          int kgi = kg + i;
          ra.s[i] = (mg < M && kgi < K)
                        ? __builtin_bit_cast(unsigned short, A[(long)mg * lda + kgi])
                        : (unsigned short)0;
        }
      }
    } else {
      int kg = kb + (tid >> 3);
      int mg = bm + ((tid & 7) << 3);
      if (kg < K && (mg + 8) <= M) {
        ra.v = *(const uint4*)(A + (long)kg * lda + mg);
      } else {
#pragma unroll
        for (int i = 0; i < 8; ++i) {
          int mgi = mg + i;
          ra.s[i] = (kg < K && mgi < M)
                        ? __builtin_bit_cast(unsigned short, A[(long)kg * lda + mgi])
                        : (unsigned short)0;
        }
      }
    }
  };
  auto loadB = [&](int kb) {
#pragma unroll
    for (int j = 0; j < 2; ++j) {
      U8& rr = j ? rb1 : rb0;
      if (BKN) {
        int kg = kb + (tid >> 3);
        int ng = bn + ((tid & 7) << 3) + j * 64;
        if (kg < K && (ng + 8) <= N) {
          rr.v = *(const uint4*)(Bm + (long)kg * ldb + ng);
        } else {
#pragma unroll
          for (int i = 0; i < 8; ++i) {
            int ngi = ng + i;
            rr.s[i] = (kg < K && ngi < N)
                          ? __builtin_bit_cast(unsigned short, Bm[(long)kg * ldb + ngi])
                          : (unsigned short)0;
          }
        }
      } else {
        int ng = bn + (tid >> 2) + j * 64;
        int kg = kb + ((tid & 3) << 3);
        if (ng < N && (kg + 8) <= K) {
          rr.v = *(const uint4*)(Bm + (long)ng * ldb + kg);
        } else {
#pragma unroll
          for (int i = 0; i < 8; ++i) {
            int kgi = kg + i;
            rr.s[i] = (ng < N && kgi < K)
                          ? __builtin_bit_cast(unsigned short, Bm[(long)ng * ldb + kgi])
                          : (unsigned short)0;
          }
        }
      }
    }
  };
  auto storeA = [&](int buf) {
    if (!AKM) {
      *(uint4*)&As[buf][tid >> 2][(tid & 3) << 3] = ra.v;
    } else {
      int kk = tid >> 3, m0 = (tid & 7) << 3;
#pragma unroll
      for (int i = 0; i < 8; ++i) As[buf][m0 + i][kk] = u2bf(ra.s[i]);
    }
  };
  auto storeB = [&](int buf) {
    if (BKN) {
      int kk = tid >> 3, n0 = (tid & 7) << 3;
#pragma unroll
      for (int i = 0; i < 8; ++i) Bs[buf][n0 + i][kk] = u2bf(rb0.s[i]);
#pragma unroll
      for (int i = 0; i < 8; ++i) Bs[buf][n0 + 64 + i][kk] = u2bf(rb1.s[i]);
    } else {
      int n0 = tid >> 2, kk = (tid & 3) << 3;
      *(uint4*)&Bs[buf][n0][kk] = rb0.v;
      *(uint4*)&Bs[buf][n0 + 64][kk] = rb1.v;
    }
  };

  v8f acc00 = {}, acc01 = {}, acc10 = {}, acc11 = {};

  loadA(0); loadB(0);
  storeA(0); storeB(0);
  __syncthreads();

  int buf = 0;
  for (int k0 = 0; k0 < K; k0 += 32) {
    const bool more = (k0 + 32) < K;
    if (more) { loadA(k0 + 32); loadB(k0 + 32); }

    // A fragments (ISA 7.12.2: 16-bit A 16x32; lane half selects K octets)
    v16bf a0, a1;
#pragma unroll
    for (int e = 0; e < 16; ++e) {
      int kk = (e < 8) ? (lh * 8 + e) : (16 + lh * 8 + (e - 8));
      a0[e] = As[buf][wm + r][kk];
      a1[e] = As[buf][wm + 16 + r][kk];
    }
    // B fragments (same K octet order as A)
    v16bf b0, b1;
#pragma unroll
    for (int e = 0; e < 16; ++e) {
      int kk = (e < 8) ? (lh * 8 + e) : (16 + lh * 8 + (e - 8));
      b0[e] = Bs[buf][wn + r][kk];
      b1[e] = Bs[buf][wn + 16 + r][kk];
    }

    acc00 = __builtin_amdgcn_wmma_f32_16x16x32_bf16(false, a0, false, b0, (short)0, acc00, false, false);
    acc01 = __builtin_amdgcn_wmma_f32_16x16x32_bf16(false, a0, false, b1, (short)0, acc01, false, false);
    acc10 = __builtin_amdgcn_wmma_f32_16x16x32_bf16(false, a1, false, b0, (short)0, acc10, false, false);
    acc11 = __builtin_amdgcn_wmma_f32_16x16x32_bf16(false, a1, false, b1, (short)0, acc11, false, false);

    if (more) { storeA(buf ^ 1); storeB(buf ^ 1); }
    __syncthreads();
    buf ^= 1;
  }

  // ---- epilogue (C layout: VGPR j -> M = j + lh*8; N = lane&15) ----
  float absSum = 0.f;
#pragma unroll
  for (int t = 0; t < 2; ++t) {
    const int ng = bn + wn + t * 16 + r;
    float bval = 0.f;
    if ((E & E_BIAS) && ng < N) bval = bias[ng];
    float colSum = 0.f;
#pragma unroll
    for (int s = 0; s < 2; ++s) {
      v8f accv = t ? (s ? acc11 : acc01) : (s ? acc10 : acc00);
#pragma unroll
      for (int j = 0; j < 8; ++j) {
        int mg = bm + wm + s * 16 + lh * 8 + j;
        bool ok = (mg < M) && (ng < N);
        float c = accv[j] + bval;
        if (E & E_RELU) c = fmaxf(c, 0.f);
        if (E & E_GELU) c = 0.5f * c * (1.f + erff(c * 0.70710678118f));
        if (E & E_ACCUM) { if (ok) c += Cf[(long)b * sC + (long)mg * ldc + ng]; }
        if (E & E_F32)   { if (ok) Cf[(long)b * sC + (long)mg * ldc + ng] = c; }
        if (E & E_BF16)  { if (ok) Cbf[(long)b * sC + (long)mg * ldc + ng] = f2bf(c); }
        if (E & E_ABS)   { if (ok) absSum += fabsf(c - ref[(long)b * sRef + (long)mg * ldc + ng]); }
        if (E & E_MEAN)  { if (ok) colSum += c; }
      }
    }
    if (E & E_MEAN) {
      if (ng < N) atomicAdd(&Cf[(long)b * sC + ng], colSum * invL);
    }
  }
  if (E & E_ABS) atomicAdd(redAcc, absSum);
}

// ---------------------------------------------------------------------------
// fp32 -> bf16 conversion, 4 elements/thread (float4 load, 8B store)
// n4 = element count / 4
// ---------------------------------------------------------------------------
__global__ void k_f2b(const float* __restrict__ in, __bf16* __restrict__ out, long n4) {
  long i = (long)blockIdx.x * 256 + threadIdx.x;
  long stride = (long)gridDim.x * 256;
  for (; i < n4; i += stride) {
    float4 v = ((const float4*)in)[i];
    union { unsigned short s[4]; uint2 u; } o;
    o.s[0] = f2bfu(v.x); o.s[1] = f2bfu(v.y);
    o.s[2] = f2bfu(v.z); o.s[3] = f2bfu(v.w);
    ((uint2*)out)[i] = o.u;
  }
}

// X[0:128] = img_cls, X[128:256] = txt_eos   (32768 float4s)
__global__ void k_copyX(const float* __restrict__ a, const float* __restrict__ b,
                        float* __restrict__ X) {
  int i = blockIdx.x * 256 + threadIdx.x;   // float4 index
  ((float4*)X)[i] = (i < 16384) ? ((const float4*)a)[i] : ((const float4*)b)[i - 16384];
}

__global__ void k_zero3(float* a) {
  if (threadIdx.x < 3) a[threadIdx.x] = 0.f;
}

// ---------------------------------------------------------------------------
// LayerNorm over D=512, one block per row, bf16 output
// ---------------------------------------------------------------------------
__global__ __launch_bounds__(256) void k_layernorm(
    const float* __restrict__ X, const float* __restrict__ sc,
    const float* __restrict__ bi, __bf16* __restrict__ out) {
  __shared__ float red[256];
  int row = blockIdx.x, tid = threadIdx.x;
  const float* x = X + (long)row * DD;
  float v0 = x[tid], v1 = x[tid + 256];
  red[tid] = v0 + v1;
  __syncthreads();
  for (int o = 128; o > 0; o >>= 1) { if (tid < o) red[tid] += red[tid + o]; __syncthreads(); }
  float mean = red[0] * (1.f / DD);
  __syncthreads();
  float d0 = v0 - mean, d1 = v1 - mean;
  red[tid] = d0 * d0 + d1 * d1;
  __syncthreads();
  for (int o = 128; o > 0; o >>= 1) { if (tid < o) red[tid] += red[tid + o]; __syncthreads(); }
  float rstd = rsqrtf(red[0] * (1.f / DD) + 1e-5f);
  out[(long)row * DD + tid]       = f2bf(d0 * rstd * sc[tid] + bi[tid]);
  out[(long)row * DD + tid + 256] = f2bf(d1 * rstd * sc[tid + 256] + bi[tid + 256]);
}

// ---------------------------------------------------------------------------
// Softmax over Li=576 per (b,t) row + s_pos/s_neg loss partials
// ---------------------------------------------------------------------------
__global__ __launch_bounds__(256) void k_softmax(
    const float* __restrict__ S, __bf16* __restrict__ Abf, float* acc) {
  __shared__ float red[256];
  long row = blockIdx.x;
  int tid = threadIdx.x;
  const float* srow = S + row * LI;
  float v[3];
#pragma unroll
  for (int i = 0; i < 3; ++i) {
    int idx = tid + i * 256;
    v[i] = (idx < LI) ? srow[idx] : 0.f;
  }
  float lm = -3.0e38f;
#pragma unroll
  for (int i = 0; i < 3; ++i) { int idx = tid + i * 256; if (idx < LI) lm = fmaxf(lm, v[i]); }
  red[tid] = lm; __syncthreads();
  for (int o = 128; o > 0; o >>= 1) { if (tid < o) red[tid] = fmaxf(red[tid], red[tid + o]); __syncthreads(); }
  float mx = red[0]; __syncthreads();
  float ln = 3.0e38f;
#pragma unroll
  for (int i = 0; i < 3; ++i) { int idx = tid + i * 256; if (idx < LI) ln = fminf(ln, v[i]); }
  red[tid] = ln; __syncthreads();
  for (int o = 128; o > 0; o >>= 1) { if (tid < o) red[tid] = fminf(red[tid], red[tid + o]); __syncthreads(); }
  float mn = red[0]; __syncthreads();
  float e[3]; float ls = 0.f;
#pragma unroll
  for (int i = 0; i < 3; ++i) {
    int idx = tid + i * 256;
    e[i] = (idx < LI) ? expf(v[i] - mx) : 0.f;
    ls += e[i];
  }
  red[tid] = ls; __syncthreads();
  for (int o = 128; o > 0; o >>= 1) { if (tid < o) red[tid] += red[tid + o]; __syncthreads(); }
  float inv = 1.f / red[0];
#pragma unroll
  for (int i = 0; i < 3; ++i) {
    int idx = tid + i * 256;
    if (idx < LI) Abf[row * LI + idx] = f2bf(e[i] * inv);
  }
  if (tid == 0) {
    float sp = 1.f / (1.f + expf(-mx));
    float sn = 1.f / (1.f + expf(-mn));
    atomicAdd(&acc[0], logf(sp + 1e-6f));
    atomicAdd(&acc[1], logf(1.f - sn + 1e-6f));
  }
}

// ---------------------------------------------------------------------------
// l2-normalize resMLP outputs + initialize f_img/f_txt with cls/eos + bias
// ---------------------------------------------------------------------------
__global__ __launch_bounds__(256) void k_l2init(
    const float* __restrict__ X, const float* __restrict__ fib,
    const float* __restrict__ ftb, float* __restrict__ out) {
  __shared__ float red[256];
  int row = blockIdx.x, tid = threadIdx.x;
  const float* x = X + (long)row * DD;
  float v0 = x[tid], v1 = x[tid + 256];
  red[tid] = v0 * v0 + v1 * v1;
  __syncthreads();
  for (int o = 128; o > 0; o >>= 1) { if (tid < o) red[tid] += red[tid + o]; __syncthreads(); }
  float inv = 1.f / fmaxf(sqrtf(red[0]), 1e-12f);
  bool img = row < BB;
  int rr = img ? row : row - BB;
  float* fdst = out + (img ? OFF_FIMG : OFF_FTXT) + (long)rr * DD;
  float* rdst = out + (img ? OFF_RESIMG : OFF_RESTXT) + (long)rr * DD;
  const float* bb = img ? fib : ftb;
  fdst[tid] = v0 + bb[tid];
  fdst[tid + 256] = v1 + bb[tid + 256];
  rdst[tid] = v0 * inv;
  rdst[tid + 256] = v1 * inv;
}

__global__ void k_final(const float* __restrict__ acc, float* __restrict__ out) {
  if (threadIdx.x == 0) {
    out[OFF_LPT]  = -(acc[0] + acc[1]) / (float)(BB * LT);
    out[OFF_LREC] = acc[2] / ((float)BB * LT * DD);
  }
}

// ---------------------------------------------------------------------------
// Host orchestration
// ---------------------------------------------------------------------------
extern "C" void kernel_launch(void* const* d_in, const int* in_sizes, int n_in,
                              void* d_out, int out_size, void* d_ws, size_t ws_size,
                              hipStream_t stream) {
  (void)in_sizes; (void)n_in; (void)out_size; (void)ws_size;
  const float* img_cls = (const float*)d_in[0];
  const float* txt_eos = (const float*)d_in[1];
  const float* img_tok = (const float*)d_in[2];
  const float* txt_tok = (const float*)d_in[3];
  const float* ln_s[2] = {(const float*)d_in[4],  (const float*)d_in[10]};
  const float* ln_b[2] = {(const float*)d_in[5],  (const float*)d_in[11]};
  const float* w1[2]   = {(const float*)d_in[6],  (const float*)d_in[12]};
  const float* b1[2]   = {(const float*)d_in[7],  (const float*)d_in[13]};
  const float* w2[2]   = {(const float*)d_in[8],  (const float*)d_in[14]};
  const float* b2[2]   = {(const float*)d_in[9],  (const float*)d_in[15]};
  const float* rec_w1  = (const float*)d_in[16];
  const float* rec_b1  = (const float*)d_in[17];
  const float* rec_w2  = (const float*)d_in[18];
  const float* rec_b2  = (const float*)d_in[19];
  const float* fi_w    = (const float*)d_in[20];
  const float* fi_b    = (const float*)d_in[21];
  const float* ft_w    = (const float*)d_in[22];
  const float* ft_b    = (const float*)d_in[23];
  float* out = (float*)d_out;

  // -------- workspace layout --------
  char* ws = (char*)d_ws;
  size_t off = 0;
  auto alloc = [&](size_t bytes) { size_t r = off; off += (bytes + 255) & ~(size_t)255; return r; };
  size_t o_X    = alloc(256L * DD * 4);
  size_t o_Xn   = alloc(256L * DD * 2);
  size_t o_H    = alloc(256L * 2048 * 2);
  size_t o_w1b[2] = {alloc(512L * 2048 * 2), alloc(512L * 2048 * 2)};
  size_t o_w2b[2] = {alloc(2048L * 512 * 2), alloc(2048L * 512 * 2)};
  size_t o_rw1  = alloc(512L * 1024 * 2);
  size_t o_rw2  = alloc(1024L * 512 * 2);
  size_t o_fiw  = alloc(1024L * 512 * 2);
  size_t o_ftw  = alloc(1024L * 512 * 2);
  size_t o_imgb = alloc((long)BB * LI * DD * 2);
  size_t o_txtb = alloc((long)BB * LT * DD * 2);
  size_t o_S    = alloc((long)BB * LT * LI * 4);
  size_t o_Abf  = alloc((long)BB * LT * LI * 2);
  size_t o_P    = alloc((long)BB * LT * DD * 4);
  size_t o_Pbf  = alloc((long)BB * LT * DD * 2);
  size_t o_Hrec = alloc((long)BB * LT * 1024 * 2);
  size_t o_ctx  = alloc((long)BB * LI * DD * 2);
  size_t o_acc  = alloc(64);

  float*  X     = (float*)(ws + o_X);
  __bf16* Xn    = (__bf16*)(ws + o_Xn);
  __bf16* Hbf   = (__bf16*)(ws + o_H);
  __bf16* w1b[2]= {(__bf16*)(ws + o_w1b[0]), (__bf16*)(ws + o_w1b[1])};
  __bf16* w2b[2]= {(__bf16*)(ws + o_w2b[0]), (__bf16*)(ws + o_w2b[1])};
  __bf16* rw1b  = (__bf16*)(ws + o_rw1);
  __bf16* rw2b  = (__bf16*)(ws + o_rw2);
  __bf16* fiwb  = (__bf16*)(ws + o_fiw);
  __bf16* ftwb  = (__bf16*)(ws + o_ftw);
  __bf16* imgb  = (__bf16*)(ws + o_imgb);
  __bf16* txtb  = (__bf16*)(ws + o_txtb);
  float*  Sf    = (float*)(ws + o_S);
  __bf16* Abf   = (__bf16*)(ws + o_Abf);
  float*  Pf    = (float*)(ws + o_P);
  __bf16* Pbf   = (__bf16*)(ws + o_Pbf);
  __bf16* Hrec  = (__bf16*)(ws + o_Hrec);
  __bf16* Ctxb  = (__bf16*)(ws + o_ctx);
  float*  acc   = (float*)(ws + o_acc);

  dim3 blk(256);
  k_zero3<<<1, 32, 0, stream>>>(acc);

  // -------- fp32 -> bf16 conversions (counts are /4) --------
  k_f2b<<<2048, blk, 0, stream>>>(img_tok, imgb, (long)BB * LI * DD / 4);
  k_f2b<<<1024, blk, 0, stream>>>(txt_tok, txtb, (long)BB * LT * DD / 4);
  for (int i = 0; i < 2; ++i) {
    k_f2b<<<512, blk, 0, stream>>>(w1[i], w1b[i], 512L * 2048 / 4);
    k_f2b<<<512, blk, 0, stream>>>(w2[i], w2b[i], 2048L * 512 / 4);
  }
  k_f2b<<<256, blk, 0, stream>>>(rec_w1, rw1b, 512L * 1024 / 4);
  k_f2b<<<256, blk, 0, stream>>>(rec_w2, rw2b, 1024L * 512 / 4);
  k_f2b<<<256, blk, 0, stream>>>(fi_w, fiwb, 1024L * 512 / 4);
  k_f2b<<<256, blk, 0, stream>>>(ft_w, ftwb, 1024L * 512 / 4);

  // -------- shared ResidualMLPs on [256, 512] (img_cls ++ txt_eos) --------
  k_copyX<<<128, blk, 0, stream>>>(img_cls, txt_eos, X);
  for (int i = 0; i < 2; ++i) {
    k_layernorm<<<256, blk, 0, stream>>>(X, ln_s[i], ln_b[i], Xn);
    // H = gelu(Xn @ w1 + b1)   [256,2048]
    gemm_wmma<(E_BF16 | E_BIAS | E_GELU), false, true>
        <<<dim3(16, 4, 1), blk, 0, stream>>>(
            Xn, 512, 0, w1b[i], 2048, 0, nullptr, Hbf, 2048, 0,
            b1[i], nullptr, 0, nullptr, 0.f, 256, 2048, 512);
    // X += H @ w2 + b2         [256,512]
    gemm_wmma<(E_F32 | E_BIAS | E_ACCUM), false, true>
        <<<dim3(4, 4, 1), blk, 0, stream>>>(
            Hbf, 2048, 0, w2b[i], 512, 0, X, nullptr, 512, 0,
            b2[i], nullptr, 0, nullptr, 0.f, 256, 512, 2048);
  }
  // res_img/res_txt + init f_img = cls + fi_b, f_txt = eos + ft_b
  k_l2init<<<256, blk, 0, stream>>>(X, fi_b, ft_b, out);

  // -------- S[b] = txt[b] @ img[b]^T   [77,576] --------
  gemm_wmma<E_F32, false, false><<<dim3(5, 2, BB), blk, 0, stream>>>(
      txtb, DD, (long)LT * DD, imgb, DD, (long)LI * DD,
      Sf, nullptr, LI, (long)LT * LI,
      nullptr, nullptr, 0, nullptr, 0.f, LT, LI, DD);

  // -------- softmax(A) + loss_pt partials --------
  k_softmax<<<BB * LT, blk, 0, stream>>>(Sf, Abf, acc);

  // -------- P_aligned[b] = A[b] @ img[b]   [77,512]  (f32 + bf16) --------
  gemm_wmma<(E_F32 | E_BF16), false, true><<<dim3(4, 2, BB), blk, 0, stream>>>(
      Abf, LI, (long)LT * LI, imgb, DD, (long)LI * DD,
      Pf, Pbf, DD, (long)LT * DD,
      nullptr, nullptr, 0, nullptr, 0.f, LT, DD, LI);

  // -------- reconstruction MLP + L1 loss --------
  // Hrec = relu(txt @ rec_w1 + rec_b1)   [9856,1024]
  gemm_wmma<(E_BF16 | E_BIAS | E_RELU), false, true>
      <<<dim3(8, 154, 1), blk, 0, stream>>>(
          txtb, DD, 0, rw1b, 1024, 0, nullptr, Hrec, 1024, 0,
          rec_b1, nullptr, 0, nullptr, 0.f, BB * LT, 1024, DD);
  // acc[2] += sum |Hrec @ rec_w2 + rec_b2 - P_aligned|
  gemm_wmma<(E_BIAS | E_ABS), false, true>
      <<<dim3(4, 154, 1), blk, 0, stream>>>(
          Hrec, 1024, 0, rw2b, DD, 0, nullptr, nullptr, DD, 0,
          rec_b2, Pf, 0, acc + 2, 0.f, BB * LT, DD, 1024);

  // -------- img_ctx[b] = A[b]^T @ txt[b]   [576,512] --------
  gemm_wmma<E_BF16, true, true><<<dim3(4, 9, BB), blk, 0, stream>>>(
      Abf, LI, (long)LT * LI, txtb, DD, (long)LT * DD,
      nullptr, Ctxb, DD, (long)LI * DD,
      nullptr, nullptr, 0, nullptr, 0.f, LI, DD, LT);

  // -------- f_img += mean_i( img@fi_w[:512] + ctx@fi_w[512:] ) --------
  gemm_wmma<E_MEAN, false, true><<<dim3(4, 9, BB), blk, 0, stream>>>(
      imgb, DD, (long)LI * DD, fiwb, DD, 0,
      out + OFF_FIMG, nullptr, DD, DD,
      nullptr, nullptr, 0, nullptr, 1.f / LI, LI, DD, DD);
  gemm_wmma<E_MEAN, false, true><<<dim3(4, 9, BB), blk, 0, stream>>>(
      Ctxb, DD, (long)LI * DD, fiwb + (long)512 * DD, DD, 0,
      out + OFF_FIMG, nullptr, DD, DD,
      nullptr, nullptr, 0, nullptr, 1.f / LI, LI, DD, DD);

  // -------- f_txt += mean_t( txt@ft_w[:512] + P@ft_w[512:] ) --------
  gemm_wmma<E_MEAN, false, true><<<dim3(4, 2, BB), blk, 0, stream>>>(
      txtb, DD, (long)LT * DD, ftwb, DD, 0,
      out + OFF_FTXT, nullptr, DD, DD,
      nullptr, nullptr, 0, nullptr, 1.f / LT, LT, DD, DD);
  gemm_wmma<E_MEAN, false, true><<<dim3(4, 2, BB), blk, 0, stream>>>(
      Pbf, DD, (long)LT * DD, ftwb + (long)512 * DD, DD, 0,
      out + OFF_FTXT, nullptr, DD, DD,
      nullptr, nullptr, 0, nullptr, 1.f / LT, LT, DD, DD);

  // -------- scalar losses --------
  k_final<<<1, 32, 0, stream>>>(acc, out);
}
